// StochasticGraphLearner_17841294147870
// MI455X (gfx1250) — matrix operand order repaired and checked by reading, not compiled, not run on hardware
//
#include <hip/hip_runtime.h>
#include <hip/hip_bf16.h>
#include <stdint.h>

// ---------------------------------------------------------------------------
// StochasticGraphLearner for MI455X (gfx1250, wave32)
//   B=4, N=1024, D=64
//   Kernel 1 (WMMA): H[4096,128] = X[4096,64] @ [W1_src | W1_dst] (+b1 on src)
//   Kernel 2 (VALU): batch-paired, i-tiled logits + symmetrize + threefry
//                    gumbel hard straight-through + adj
// ---------------------------------------------------------------------------

#define BATCH 4
#define NN    1024
#define DD    64
#define ROWS  (BATCH * NN)          // 4096
#define ADJ_ELEMS (BATCH * NN * NN) // 4194304
#define GUMBEL_HALF 4194304u        // (B*N*N*2)/2 == 2*N*N*2 / ... = n/2
#define TI    4                     // i-rows per block in kernel 2

typedef __attribute__((ext_vector_type(16))) _Float16 v16h;
typedef __attribute__((ext_vector_type(8)))  float    v8f;

// ---------------------------------------------------------------------------
// Kernel 1: projection GEMM with v_wmma_f32_16x16x32_f16.
// Grid: 256 blocks (one 16-row M tile each) x 256 threads (8 waves; wave w
// owns output columns [16w, 16w+16) of the 128 columns).
// ---------------------------------------------------------------------------
__global__ void __launch_bounds__(256)
sgl_proj_wmma(const float* __restrict__ X,    // [4096, 64]
              const float* __restrict__ W1,   // [128, 64] (in-major)
              const float* __restrict__ b1,   // [64]
              float* __restrict__ H)          // [4096, 128] out
{
    const int lane = threadIdx.x & 31;
    const int wave = threadIdx.x >> 5;
    const int m0   = blockIdx.x * 16;
    const int n0   = wave * 16;

    const int m   = lane & 15;
    const int n   = n0 + (lane & 15);
    const int lhi = lane >> 4;

    v8f c = {};
#pragma unroll
    for (int kk = 0; kk < DD; kk += 32) {
        v16h a, bm;
#pragma unroll
        for (int h = 0; h < 16; ++h) {
            // A 16x32 f16 layout: lane group K-base 0/8; halves 0..7 ->
            // K base+0..7, halves 8..15 -> K base+16..23
            const int ka = kk + lhi * 8 + (h & 7) + ((h >> 3) << 4);
            a[h] = (_Float16)X[(m0 + m) * DD + ka];
            // B 32x16 f16 layout: same K order as A (base 0/8; halves 8..15 -> +16)
            const int kb = kk + lhi * 8 + (h & 7) + ((h >> 3) << 4);
            const float bv = (n < DD) ? W1[kb * DD + n]
                                      : W1[(DD + kb) * DD + (n - DD)];
            bm[h] = (_Float16)bv;
        }
        c = __builtin_amdgcn_wmma_f32_16x16x32_f16(
                false, a, false, bm, (short)0, c, false, false);
    }

    const float badd = (n < DD) ? b1[n] : 0.0f;   // fold b1 into src half
#pragma unroll
    for (int v = 0; v < 8; ++v) {
        const int mr = v + (lhi << 3);
        H[(size_t)(m0 + mr) * 128 + n] = c[v] + badd;
    }
}

// ---------------------------------------------------------------------------
// JAX threefry2x32 with key = threefry_seed(42) = [0, 42]
// ---------------------------------------------------------------------------
__device__ __forceinline__ void threefry2x32_42(uint32_t x0, uint32_t x1,
                                                uint32_t& o0, uint32_t& o1)
{
    const uint32_t ks0 = 0u, ks1 = 42u;
    const uint32_t ks2 = ks0 ^ ks1 ^ 0x1BD11BDAu;
    const uint32_t ks[3] = {ks0, ks1, ks2};
    const int rot[2][4] = {{13, 15, 26, 6}, {17, 29, 16, 24}};
    x0 += ks0; x1 += ks1;
#pragma unroll
    for (int g = 0; g < 5; ++g) {
        const int* r = rot[g & 1];
#pragma unroll
        for (int q = 0; q < 4; ++q) {
            x0 += x1;
            x1 = (x1 << r[q]) | (x1 >> (32 - r[q]));
            x1 ^= x0;
        }
        x0 += ks[(g + 1) % 3];
        x1 += ks[(g + 2) % 3] + (uint32_t)(g + 1);
    }
    o0 = x0; o1 = x1;
}

__device__ __forceinline__ float bits_to_gumbel(uint32_t bits)
{
    float u = __uint_as_float(0x3F800000u | (bits >> 9)) - 1.0f; // [0,1)
    const float tiny = 1.175494350822288e-38f;
    u = u * (1.0f - tiny) + tiny;
    u = fmaxf(tiny, u);
    return -__logf(-__logf(u));
}

// ---------------------------------------------------------------------------
// Kernel 2: block = (batch-pair bp, i-tile of 4 rows); thread covers 4 j's.
// Batch pairing {bp, bp+2}: JAX's half-split threefry means one call yields
// bits for element idx (out0) and element idx + 2*N*N (out1) -- the same
// (i,j) in batch bp+2 -- so each threefry call pair serves TWO outputs.
// Computes both (i,j) and (j,i) sums so symmetrization stays local.
// ---------------------------------------------------------------------------
__global__ void __launch_bounds__(256)
sgl_edges(const float* __restrict__ H,   // [4096, 128] (src half has +b1)
          const float* __restrict__ W2,  // [64]
          const float* __restrict__ b2,  // [1]
          float* __restrict__ out)       // [adj(4M) | logits(4M)]
{
    const int bp = blockIdx.x >> 8;          // 0..1 -> batches {bp, bp+2}
    const int i0 = (blockIdx.x & 255) * TI;  // first i row of this tile
    const int tid = threadIdx.x;

    __shared__ alignas(16) float sS[2][TI][DD];  // h_src[i]+b1, per batch
    __shared__ alignas(16) float tS[2][TI][DD];  // h_dst[i],    per batch
    __shared__ alignas(16) float w2s[DD];

    // Stage 2 batches x TI rows x 128 floats = 1024 floats with 256 threads.
#pragma unroll
    for (int q = 0; q < 2 * TI * 128 / 256; ++q) {
        const int fidx = tid + q * 256;
        const int row  = fidx >> 7;           // 0..7
        const int s    = row >> 2;            // batch-pair half
        const int ii   = row & (TI - 1);
        const int off  = fidx & 127;
        const float v = H[(size_t)((bp + 2 * s) * NN + i0 + ii) * 128 + off];
        if (off < DD) sS[s][ii][off] = v;
        else          tS[s][ii][off - DD] = v;
    }
    if (tid < DD) w2s[tid] = W2[tid];
    __syncthreads();

    const float bias2 = b2[0];
    const float* __restrict__ Hb0 = H + (size_t)bp * NN * 128;        // batch bp
    const float* __restrict__ Hb1 = H + (size_t)(bp + 2) * NN * 128;  // batch bp+2

#pragma unroll
    for (int r = 0; r < 4; ++r) {
        const int j = tid + (r << 8);
        const float4* __restrict__ hj0 = (const float4*)(Hb0 + (size_t)j * 128);
        const float4* __restrict__ hj1 = (const float4*)(Hb1 + (size_t)j * 128);
        if (r < 3) {  // prefetch next j rows -> global_prefetch_b8
            __builtin_prefetch(Hb0 + (size_t)(j + 256) * 128, 0, 0);
            __builtin_prefetch(Hb1 + (size_t)(j + 256) * 128, 0, 0);
        }

        float aij[2][TI], aji[2][TI];
#pragma unroll
        for (int s = 0; s < 2; ++s)
#pragma unroll
            for (int ii = 0; ii < TI; ++ii) { aij[s][ii] = 0.f; aji[s][ii] = 0.f; }

#pragma unroll
        for (int d4 = 0; d4 < DD / 4; ++d4) {
            const float4 sj0 = hj0[d4];           // h_src[j]+b1, batch bp
            const float4 tj0 = hj0[DD / 4 + d4];  // h_dst[j],    batch bp
            const float4 sj1 = hj1[d4];
            const float4 tj1 = hj1[DD / 4 + d4];
            const float4 w   = ((const float4*)w2s)[d4];
#pragma unroll
            for (int s = 0; s < 2; ++s) {
                const float4 sj = s ? sj1 : sj0;
                const float4 tj = s ? tj1 : tj0;
#pragma unroll
                for (int ii = 0; ii < TI; ++ii) {
                    const float4 si = ((const float4*)sS[s][ii])[d4];
                    const float4 ti = ((const float4*)tS[s][ii])[d4];
                    float a = aij[s][ii], c = aji[s][ii];
                    a = fmaf(fmaxf(si.x + tj.x, 0.f), w.x, a);
                    a = fmaf(fmaxf(si.y + tj.y, 0.f), w.y, a);
                    a = fmaf(fmaxf(si.z + tj.z, 0.f), w.z, a);
                    a = fmaf(fmaxf(si.w + tj.w, 0.f), w.w, a);
                    c = fmaf(fmaxf(sj.x + ti.x, 0.f), w.x, c);
                    c = fmaf(fmaxf(sj.y + ti.y, 0.f), w.y, c);
                    c = fmaf(fmaxf(sj.z + ti.z, 0.f), w.z, c);
                    c = fmaf(fmaxf(sj.w + ti.w, 0.f), w.w, c);
                    aij[s][ii] = a; aji[s][ii] = c;
                }
            }
        }

#pragma unroll
        for (int ii = 0; ii < TI; ++ii) {
            const int i = i0 + ii;
            const float sym0 = 0.5f * (aij[0][ii] + aji[0][ii]) + bias2;
            const float sym1 = 0.5f * (aij[1][ii] + aji[1][ii]) + bias2;

            // idx0 in batch bp (< 2*N*N), idx1 = idx0 + 2*N*N (batch bp+2).
            const uint32_t idx0 = ((uint32_t)(bp * NN + i) * NN) + (uint32_t)j;
            const uint32_t idx1 = idx0 + 2u * NN * NN;
            const uint32_t base = idx0 * 2u;      // < GUMBEL_HALF by construction

            uint32_t g0b0, g0b1, g1b0, g1b1;
            threefry2x32_42(base,      base + GUMBEL_HALF,      g0b0, g0b1);
            threefry2x32_42(base + 1u, base + 1u + GUMBEL_HALF, g1b0, g1b1);
            const float g0_0 = bits_to_gumbel(g0b0);  // elem idx0, slot 0
            const float g1_0 = bits_to_gumbel(g1b0);  // elem idx0, slot 1
            const float g0_1 = bits_to_gumbel(g0b1);  // elem idx1, slot 0
            const float g1_1 = bits_to_gumbel(g1b1);  // elem idx1, slot 1

            const float adj0 = ((sym0 + g0_0) >= g1_0 && (i != j)) ? 1.0f : 0.0f;
            const float adj1 = ((sym1 + g0_1) >= g1_1 && (i != j)) ? 1.0f : 0.0f;

            out[idx0] = adj0;
            out[idx1] = adj1;
            out[(size_t)ADJ_ELEMS + idx0] = sym0;
            out[(size_t)ADJ_ELEMS + idx1] = sym1;
        }
    }
}

// ---------------------------------------------------------------------------
extern "C" void kernel_launch(void* const* d_in, const int* in_sizes, int n_in,
                              void* d_out, int out_size, void* d_ws, size_t ws_size,
                              hipStream_t stream)
{
    const float* X  = (const float*)d_in[0];  // [4,1024,64]
    const float* W1 = (const float*)d_in[1];  // [128,64]
    const float* b1 = (const float*)d_in[2];  // [64]
    const float* W2 = (const float*)d_in[3];  // [64,1]
    const float* b2 = (const float*)d_in[4];  // [1]
    float* out = (float*)d_out;               // [adj(4M) | logits(4M)]
    float* H   = (float*)d_ws;                // [4096,128] = 2 MB scratch

    // Projection GEMM: 256 M-tiles x 8 N-tiles, 8 waves/block.
    sgl_proj_wmma<<<ROWS / 16, 256, 0, stream>>>(X, W1, b1, H);
    // Edge pass: 2 batch-pairs x 256 i-tiles = 512 blocks x 256 threads.
    sgl_edges<<<2 * (NN / TI), 256, 0, stream>>>(H, W2, b2, out);
}
